// RelationalAttentionLayer_5815385718827
// MI455X (gfx1250) — matrix operand order repaired and checked by reading, not compiled, not run on hardware
//
#include <hip/hip_runtime.h>
#include <hip/hip_bf16.h>

typedef __attribute__((ext_vector_type(16))) __bf16 v16bf;
typedef __attribute__((ext_vector_type(8)))  float  v8f;

#define NN   20000      // nodes
#define NE   320000     // edges
#define DIMK 128        // model dim (K of GEMM1)
#define HD   256        // H*D
#define NWC  1792       // 7*256 fused output columns of GEMM1
#define EPSV 1e-10f

union FragU  { v16bf v; uint4 u[2]; };
union Half8  { uint4 u; __hip_bfloat16 h[8]; };

// ---------------- utility kernels ----------------

__global__ void zero_f32_kernel(float* __restrict__ p, int n) {
    int i = blockIdx.x * blockDim.x + threadIdx.x;
    int stride = gridDim.x * blockDim.x;
    for (; i < n; i += stride) p[i] = 0.0f;
}

__global__ void cvt_f32_bf16_kernel(const float* __restrict__ x,
                                    __hip_bfloat16* __restrict__ y, int n) {
    int i = blockIdx.x * blockDim.x + threadIdx.x;
    if (i < n) y[i] = __float2bfloat16(x[i]);
}

// Pack WQ | WK | WV -> W_all bf16 [1792, 128] row-major
__global__ void pack_weights_kernel(const float* __restrict__ WQ,
                                    const float* __restrict__ WK,
                                    const float* __restrict__ WV,
                                    __hip_bfloat16* __restrict__ Wall) {
    int i = blockIdx.x * blockDim.x + threadIdx.x;
    const int total = NWC * DIMK;
    if (i >= total) return;
    float v;
    if (i < 256 * DIMK)            v = WQ[i];
    else if (i < 1024 * DIMK)      v = WK[i - 256 * DIMK];
    else                           v = WV[i - 1024 * DIMK];
    Wall[i] = __float2bfloat16(v);
}

// -------- WMMA GEMM: C[M,Ncols] = X[M,K] * W[Ncols,K]^T --------
// One wave32 per 16x64 strip: A fragment loaded once per K-step,
// reused across 4 WMMAs (4 N-tiles, 4 accumulators).  K % 32 == 0,
// Ncols % 64 == 0.  bf16-output variant (GEMM1).

__global__ void gemm_wmma_bf16out_kernel(const __hip_bfloat16* __restrict__ X,
                                         const __hip_bfloat16* __restrict__ W,
                                         __hip_bfloat16* __restrict__ C,
                                         int K, int Ncols, int totalTasks) {
    const int groupsN = Ncols >> 6;                      // 64-wide column groups
    int task = blockIdx.x * blockDim.y + threadIdx.y;    // wave-uniform
    if (task >= totalTasks) return;
    int tM = task / groupsN;
    int tG = task - tM * groupsN;

    int lane = threadIdx.x;
    int half = lane >> 4;        // 0: lanes 0-15, 1: lanes 16-31
    int l16  = lane & 15;

    const __hip_bfloat16* Arow = X + (size_t)((tM << 4) + l16) * K;
    const __hip_bfloat16* Brow = W + (size_t)((tG << 6) + l16) * K;
    const size_t bStep = (size_t)16 * K;                 // next 16x16 B tile

    v8f acc0 = {}, acc1 = {}, acc2 = {}, acc3 = {};
    #pragma unroll 4
    for (int kk = 0; kk < K; kk += 32) {
        FragU a;
        int ka = kk + (half ? 16 : 0);     // A lanes<16: K 0-15; lanes>=16: K 16-31
        a.u[0] = *(const uint4*)(Arow + ka);
        a.u[1] = *(const uint4*)(Arow + ka + 8);
        int kb = kk + (half ? 16 : 0);     // B lanes<16: K 0-15; lanes>=16: K 16-31
        FragU b0, b1, b2, b3;
        b0.u[0] = *(const uint4*)(Brow + kb);
        b0.u[1] = *(const uint4*)(Brow + kb + 8);
        b1.u[0] = *(const uint4*)(Brow + bStep + kb);
        b1.u[1] = *(const uint4*)(Brow + bStep + kb + 8);
        b2.u[0] = *(const uint4*)(Brow + 2 * bStep + kb);
        b2.u[1] = *(const uint4*)(Brow + 2 * bStep + kb + 8);
        b3.u[0] = *(const uint4*)(Brow + 3 * bStep + kb);
        b3.u[1] = *(const uint4*)(Brow + 3 * bStep + kb + 8);
        acc0 = __builtin_amdgcn_wmma_f32_16x16x32_bf16(false, a.v, false, b0.v, (short)0, acc0, false, false);
        acc1 = __builtin_amdgcn_wmma_f32_16x16x32_bf16(false, a.v, false, b1.v, (short)0, acc1, false, false);
        acc2 = __builtin_amdgcn_wmma_f32_16x16x32_bf16(false, a.v, false, b2.v, (short)0, acc2, false, false);
        acc3 = __builtin_amdgcn_wmma_f32_16x16x32_bf16(false, a.v, false, b3.v, (short)0, acc3, false, false);
    }
    // C frag: VGPR j -> M = j + 8*half, N = l16 (+16 per tile)
    int mBase = (tM << 4) + (half << 3);
    int nCol  = (tG << 6) + l16;
    #pragma unroll
    for (int j = 0; j < 8; ++j) {
        __hip_bfloat16* cr = C + (size_t)(mBase + j) * Ncols + nCol;
        cr[0]  = __float2bfloat16(acc0[j]);
        cr[16] = __float2bfloat16(acc1[j]);
        cr[32] = __float2bfloat16(acc2[j]);
        cr[48] = __float2bfloat16(acc3[j]);
    }
}

// f32-output variant (GEMM2: out = z_bf16 * WO^T)
__global__ void gemm_wmma_f32out_kernel(const __hip_bfloat16* __restrict__ X,
                                        const __hip_bfloat16* __restrict__ W,
                                        float* __restrict__ C,
                                        int K, int Ncols, int totalTasks) {
    const int groupsN = Ncols >> 6;
    int task = blockIdx.x * blockDim.y + threadIdx.y;
    if (task >= totalTasks) return;
    int tM = task / groupsN;
    int tG = task - tM * groupsN;

    int lane = threadIdx.x;
    int half = lane >> 4;
    int l16  = lane & 15;

    const __hip_bfloat16* Arow = X + (size_t)((tM << 4) + l16) * K;
    const __hip_bfloat16* Brow = W + (size_t)((tG << 6) + l16) * K;
    const size_t bStep = (size_t)16 * K;

    v8f acc0 = {}, acc1 = {}, acc2 = {}, acc3 = {};
    #pragma unroll 4
    for (int kk = 0; kk < K; kk += 32) {
        FragU a;
        int ka = kk + (half ? 16 : 0);
        a.u[0] = *(const uint4*)(Arow + ka);
        a.u[1] = *(const uint4*)(Arow + ka + 8);
        int kb = kk + (half ? 16 : 0);
        FragU b0, b1, b2, b3;
        b0.u[0] = *(const uint4*)(Brow + kb);
        b0.u[1] = *(const uint4*)(Brow + kb + 8);
        b1.u[0] = *(const uint4*)(Brow + bStep + kb);
        b1.u[1] = *(const uint4*)(Brow + bStep + kb + 8);
        b2.u[0] = *(const uint4*)(Brow + 2 * bStep + kb);
        b2.u[1] = *(const uint4*)(Brow + 2 * bStep + kb + 8);
        b3.u[0] = *(const uint4*)(Brow + 3 * bStep + kb);
        b3.u[1] = *(const uint4*)(Brow + 3 * bStep + kb + 8);
        acc0 = __builtin_amdgcn_wmma_f32_16x16x32_bf16(false, a.v, false, b0.v, (short)0, acc0, false, false);
        acc1 = __builtin_amdgcn_wmma_f32_16x16x32_bf16(false, a.v, false, b1.v, (short)0, acc1, false, false);
        acc2 = __builtin_amdgcn_wmma_f32_16x16x32_bf16(false, a.v, false, b2.v, (short)0, acc2, false, false);
        acc3 = __builtin_amdgcn_wmma_f32_16x16x32_bf16(false, a.v, false, b3.v, (short)0, acc3, false, false);
    }
    int mBase = (tM << 4) + (half << 3);
    int nCol  = (tG << 6) + l16;
    #pragma unroll
    for (int j = 0; j < 8; ++j) {
        float* cr = C + (size_t)(mBase + j) * Ncols + nCol;
        cr[0]  = acc0[j];
        cr[16] = acc1[j];
        cr[32] = acc2[j];
        cr[48] = acc3[j];
    }
}

// ---------------- edge kernels (wave32 per edge) ----------------
// C layout per node row (NWC=1792 bf16): [0,256)=q, [256+r*256)=k_r, [1024+r*256)=v_r
// Lanes 0-15 own head 0 (8 contiguous halves each), lanes 16-31 own head 1.

__global__ void edge_score_kernel(const __hip_bfloat16* __restrict__ C,
                                  const int* __restrict__ esrc,
                                  const int* __restrict__ edst,
                                  const int* __restrict__ erel,
                                  float* __restrict__ numer,
                                  float* __restrict__ denom) {
    int e = blockIdx.x * (blockDim.x >> 5) + (threadIdx.x >> 5);
    if (e >= NE) return;
    int lane = threadIdx.x & 31;
    int head = lane >> 4;
    int idx  = head * 128 + (lane & 15) * 8;     // element offset within [0,256)
    int s = esrc[e], d = edst[e], r = erel[e];

    Half8 q, k;
    q.u = *(const uint4*)(C + (size_t)d * NWC + idx);
    k.u = *(const uint4*)(C + (size_t)s * NWC + 256 + r * 256 + idx);

    float acc = 0.f;
    #pragma unroll
    for (int i = 0; i < 8; ++i)
        acc += __bfloat162float(q.h[i]) * __bfloat162float(k.h[i]);

    // reduce within each 16-lane head group
    #pragma unroll
    for (int off = 8; off; off >>= 1)
        acc += __shfl_xor(acc, off, 32);

    if ((lane & 15) == 0) {                      // lane 0 -> head 0, lane 16 -> head 1
        const float inv_scale = 0.0625f;         // 1/sqrt(256)
        float a = fmaxf(acc * inv_scale, 0.f);
        a = a * a + EPSV;
        numer[2 * e + head] = a;
        atomicAdd(&denom[2 * d + head], a);
    }
}

__global__ void edge_aggregate_kernel(const __hip_bfloat16* __restrict__ C,
                                      const int* __restrict__ esrc,
                                      const int* __restrict__ edst,
                                      const int* __restrict__ erel,
                                      const float* __restrict__ numer,
                                      const float* __restrict__ denom,
                                      float* __restrict__ z) {
    int e = blockIdx.x * (blockDim.x >> 5) + (threadIdx.x >> 5);
    if (e >= NE) return;
    int lane = threadIdx.x & 31;
    int head = lane >> 4;
    int idx  = head * 128 + (lane & 15) * 8;
    int s = esrc[e], d = edst[e], r = erel[e];

    float w = numer[2 * e + head] / denom[2 * d + head];

    Half8 v;
    v.u = *(const uint4*)(C + (size_t)s * NWC + 1024 + r * 256 + idx);
    float* zr = z + (size_t)d * HD + idx;
    #pragma unroll
    for (int i = 0; i < 8; ++i)
        atomicAdd(&zr[i], w * __bfloat162float(v.h[i]));
}

// ---------------- launch ----------------

extern "C" void kernel_launch(void* const* d_in, const int* in_sizes, int n_in,
                              void* d_out, int out_size, void* d_ws, size_t ws_size,
                              hipStream_t stream) {
    const float* node_feature = (const float*)d_in[0];   // [N,128]
    const float* WQ  = (const float*)d_in[1];            // [256,128]
    const float* WK  = (const float*)d_in[2];            // [3,256,128]
    const float* WV  = (const float*)d_in[3];            // [3,256,128]
    const float* WO  = (const float*)d_in[4];            // [128,256]
    const int* esrc  = (const int*)d_in[5];
    const int* edst  = (const int*)d_in[6];
    const int* erel  = (const int*)d_in[7];
    float* out = (float*)d_out;                          // [N,128] f32

    // workspace carve-up (256B aligned)
    char* ws = (char*)d_ws;
    size_t off = 0;
    auto carve = [&](size_t bytes) -> char* {
        char* p = ws + off;
        off = (off + bytes + 255) & ~(size_t)255;
        return p;
    };
    __hip_bfloat16* Xbf   = (__hip_bfloat16*)carve((size_t)NN * DIMK * 2);      // 5.1 MB
    __hip_bfloat16* Wall  = (__hip_bfloat16*)carve((size_t)NWC * DIMK * 2);     // 459 KB
    __hip_bfloat16* WObf  = (__hip_bfloat16*)carve((size_t)DIMK * HD * 2);      // 64 KB
    __hip_bfloat16* Cbf   = (__hip_bfloat16*)carve((size_t)NN * NWC * 2);       // 71.7 MB
    float* numer          = (float*)carve((size_t)NE * 2 * 4);                  // 2.6 MB
    float* denom          = (float*)carve((size_t)NN * 2 * 4);                  // 160 KB (256-mult)
    float* z              = (float*)carve((size_t)NN * HD * 4);                 // 20.5 MB (contig after denom)
    __hip_bfloat16* zbf   = (__hip_bfloat16*)carve((size_t)NN * HD * 2);        // 10.2 MB
    (void)ws_size;

    // 1. convert inputs to bf16
    cvt_f32_bf16_kernel<<<(NN * DIMK + 255) / 256, 256, 0, stream>>>(node_feature, Xbf, NN * DIMK);
    pack_weights_kernel<<<(NWC * DIMK + 255) / 256, 256, 0, stream>>>(WQ, WK, WV, Wall);
    cvt_f32_bf16_kernel<<<(DIMK * HD + 255) / 256, 256, 0, stream>>>(WO, WObf, DIMK * HD);

    // 2. zero denom + z (contiguous region: NN*2 + NN*256 floats)
    zero_f32_kernel<<<4096, 256, 0, stream>>>(denom, NN * 2 + NN * HD);

    // 3. fused GEMM1: C[N,1792] = X[N,128] * Wall[1792,128]^T   (WMMA bf16)
    {
        const int totalTasks = (NN / 16) * (NWC / 64);   // 1250*28 = 35000
        dim3 blk(32, 4);
        gemm_wmma_bf16out_kernel<<<totalTasks / 4, blk, 0, stream>>>(
            Xbf, Wall, Cbf, DIMK, NWC, totalTasks);
    }

    // 4. edge scores + denominator segment-sum
    edge_score_kernel<<<NE / 8, 256, 0, stream>>>(Cbf, esrc, edst, erel, numer, denom);

    // 5. normalized weighted scatter-add into z
    edge_aggregate_kernel<<<NE / 8, 256, 0, stream>>>(Cbf, esrc, edst, erel, numer, denom, z);

    // 6. z -> bf16
    cvt_f32_bf16_kernel<<<(NN * HD + 255) / 256, 256, 0, stream>>>(z, zbf, NN * HD);

    // 7. GEMM2: out[N,128] = z[N,256] * WO[128,256]^T   (WMMA bf16, f32 out)
    {
        const int totalTasks = (NN / 16) * (DIMK / 64);  // 1250*2 = 2500
        dim3 blk(32, 4);
        gemm_wmma_f32out_kernel<<<totalTasks / 4, blk, 0, stream>>>(
            zbf, WObf, out, HD, DIMK, totalTasks);
    }
    (void)in_sizes; (void)n_in; (void)out_size;
}
